// DomainContext_14035953123328
// MI455X (gfx1250) — matrix operand order repaired and checked by reading, not compiled, not run on hardware
//
#include <hip/hip_runtime.h>
#include <hip/hip_bf16.h>
#include <stdint.h>

// ---------------------------------------------------------------------------
// Sizes (fixed by the reference)
// ---------------------------------------------------------------------------
#define B_    128
#define T_    1024
#define STATE 450
#define ACT_  59
#define DIN   512
#define HID   512
#define G3H   1536          // 3*HID
#define CTX   128
#define NT_IG (G3H / 16)    // 96 n-tiles
#define KT_   (HID / 32)    // 16 k-tiles of 32
#define XSTR  261           // LDS row stride in dwords (522 bf16) - conflict-free
#define HGSTR 1540          // hg LDS row stride in floats

typedef __attribute__((ext_vector_type(16))) __bf16        v16bf;
typedef __attribute__((ext_vector_type(8)))  float         v8f;
typedef __attribute__((ext_vector_type(8)))  unsigned int  u32x8;

__device__ __forceinline__ unsigned short f2bf(float f) {
    unsigned u = __float_as_uint(f);
    unsigned r = u + 0x7FFFu + ((u >> 16) & 1u);   // round-to-nearest-even
    return (unsigned short)(r >> 16);
}
__device__ __forceinline__ unsigned pack2bf(float lo, float hi) {
    return (unsigned)f2bf(lo) | ((unsigned)f2bf(hi) << 16);
}
__device__ __forceinline__ v16bf as16bf(u32x8 u) { return __builtin_bit_cast(v16bf, u); }

// two coalesced b128 loads -> one B operand (8 dwords per lane)
__device__ __forceinline__ u32x8 loadB16(const uint4* __restrict__ p) {
    uint4 lo = p[0], hi = p[1];
    u32x8 b;
    b[0]=lo.x; b[1]=lo.y; b[2]=lo.z; b[3]=lo.w;
    b[4]=hi.x; b[5]=hi.y; b[6]=hi.z; b[7]=hi.w;
    return b;
}

__device__ __forceinline__ float fsigmoid(float x) {
    return 1.0f / (1.0f + __expf(-x));
}
__device__ __forceinline__ float ftanh(float x) {
    x = fminf(fmaxf(x, -15.0f), 15.0f);
    float e = __expf(2.0f * x);
    return (e - 1.0f) / (e + 1.0f);
}

// ---------------------------------------------------------------------------
// 1) enc_wT[d][h] = enc_w[h][d]
// ---------------------------------------------------------------------------
__global__ void k_transpose_encw(const float* __restrict__ enc_w, float* __restrict__ enc_wT) {
    int i = blockIdx.x * blockDim.x + threadIdx.x;       // 0 .. 512*512-1
    int d = i >> 9, h = i & 511;
    enc_wT[(size_t)d * 512 + h] = enc_w[(size_t)h * 512 + d];
}

// ---------------------------------------------------------------------------
// 2) w_comb = w_ih @ enc_w  computed f32, emitted directly in packed bf16
//    WMMA *B-operand* tile layout:
//      tile(nt,kt): 256 dwords; lane L, vgpr v -> dword  tile*256 + L*8 + v
//      element:  n = nt*16 + (L&15),  k = kt*32 + (v>=4?16:0) + (L>=16?8:0) + 2(v&3) (+0/+1 in dword)
// ---------------------------------------------------------------------------
__global__ void k_pack_wcomb(const float* __restrict__ w_ih, const float* __restrict__ enc_wT,
                             unsigned* __restrict__ wcombp) {
    int p    = blockIdx.x * blockDim.x + threadIdx.x;    // 0 .. 96*16*256-1
    int tile = p >> 8, w = p & 255;
    int L = w >> 3, v = w & 7;
    int nt = tile >> 4, kt = tile & 15;
    int n  = nt * 16 + (L & 15);
    int k  = kt * 32 + ((v & 4) << 2) + ((L >> 4) << 3) + ((v & 3) << 1);
    const float* wr = w_ih  + (size_t)n * 512;
    const float* e0 = enc_wT + (size_t)k * 512;
    const float* e1 = e0 + 512;
    float s0 = 0.f, s1 = 0.f;
    for (int j = 0; j < 512; ++j) {
        float wi = wr[j];
        s0 += wi * e0[j];
        s1 += wi * e1[j];
    }
    wcombp[p] = pack2bf(s0, s1);
}

// ---------------------------------------------------------------------------
// 3) b_comb[g] = b_ih[g] + w_ih[g,:] . enc_b
// ---------------------------------------------------------------------------
__global__ void k_bias_comb(const float* __restrict__ w_ih, const float* __restrict__ enc_b,
                            const float* __restrict__ b_ih, float* __restrict__ b_comb) {
    int g = blockIdx.x * blockDim.x + threadIdx.x;       // 0..1535
    const float* wr = w_ih + (size_t)g * 512;
    float s = b_ih[g];
    for (int j = 0; j < 512; ++j) s += wr[j] * enc_b[j];
    b_comb[g] = s;
}

// ---------------------------------------------------------------------------
// 4) pack w_hh (f32 [1536,512]) into the same bf16 B-operand tile layout
// ---------------------------------------------------------------------------
__global__ void k_pack_whh(const float* __restrict__ w_hh, unsigned* __restrict__ whhp) {
    int p    = blockIdx.x * blockDim.x + threadIdx.x;
    int tile = p >> 8, w = p & 255;
    int L = w >> 3, v = w & 7;
    int nt = tile >> 4, kt = tile & 15;
    int n  = nt * 16 + (L & 15);
    int k  = kt * 32 + ((v & 4) << 2) + ((L >> 4) << 3) + ((v & 3) << 1);
    const float* src = w_hh + (size_t)n * 512 + k;
    whhp[p] = pack2bf(src[0], src[1]);
}

// ---------------------------------------------------------------------------
// 5) IG GEMM: ig[t,b,g] = X[b*T+t,:] . w_comb[g,:] + b_comb[g]
//    Grid: 2048 WGs x 512 thr. WG = 64 rows; 16 waves = 4(m) x 4(n);
//    wave: m-tile fixed, 24 n-tiles in 6 groups of 4; batched B loads per kt.
// ---------------------------------------------------------------------------
__global__ void __launch_bounds__(512)
k_ig_gemm(const float* __restrict__ obs, const float* __restrict__ rew,
          const float* __restrict__ cst, const float* __restrict__ ter,
          const float* __restrict__ act,
          const uint4* __restrict__ wcombp4, const float* __restrict__ b_comb,
          float* __restrict__ ig) {
    extern __shared__ char smem[];
    unsigned short* xh = (unsigned short*)smem;          // 64 rows x 522 bf16
    const unsigned* xu = (const unsigned*)smem;

    const int tid  = threadIdx.x;
    const int lane = tid & 31;
    const int wave = tid >> 5;
    const int mw   = wave >> 2;                          // 0..3
    const int nw   = wave & 3;                           // 0..3
    const int wg   = blockIdx.x;

    // ---- stage 64 rows of X (concat + f32->bf16), streaming (NT) loads ----
    for (int it = 0; it < 64; ++it) {
        int idx = it * 512 + tid;                        // 0..32767
        int r = idx >> 9, c = idx & 511;
        int m = wg * 64 + r;
        float vv;
        if (c < STATE)            vv = __builtin_nontemporal_load(obs + (size_t)m * STATE + c);
        else if (c == STATE)      vv = __builtin_nontemporal_load(rew + m);
        else if (c == STATE + 1)  vv = __builtin_nontemporal_load(cst + m);
        else if (c == STATE + 2)  vv = __builtin_nontemporal_load(ter + m);
        else                      vv = __builtin_nontemporal_load(act + (size_t)m * ACT_ + (c - (STATE + 3)));
        xh[r * 522 + c] = f2bf(vv);
    }
    __syncthreads();

    const int mrow = mw * 16 + (lane & 15);
    const int khi  = ((lane >> 4) << 3);                 // +8 for upper half-wave (A layout)
    const int mlo  = mw * 16 + ((lane >> 4) << 3);       // C layout row base

    for (int g = 0; g < 6; ++g) {
        int nt0 = nw * 24 + g * 4;
        const uint4* bbase = wcombp4 + (size_t)nt0 * KT_ * 64 + lane * 2;
        v8f acc[4];
#pragma unroll
        for (int i = 0; i < 4; ++i) acc[i] = (v8f){0.f,0.f,0.f,0.f,0.f,0.f,0.f,0.f};

#pragma unroll
        for (int kt = 0; kt < KT_; ++kt) {
            // batch all B loads for this kt, then A, then 4 wmma
            u32x8 bu[4];
#pragma unroll
            for (int i = 0; i < 4; ++i) bu[i] = loadB16(bbase + (i * KT_ + kt) * 64);
            u32x8 au;
#pragma unroll
            for (int v = 0; v < 8; ++v) {
                int k = kt * 32 + ((v & 4) << 2) + khi + ((v & 3) << 1);
                au[v] = xu[mrow * XSTR + (k >> 1)];
            }
            v16bf A = as16bf(au);
#pragma unroll
            for (int i = 0; i < 4; ++i)
                acc[i] = __builtin_amdgcn_wmma_f32_16x16x32_bf16(
                    false, A, false, as16bf(bu[i]), (short)0, acc[i], false, false);
        }
        // store: D layout -> element (m = r + (lane>=16?8:0), n = lane&15), NT stores
#pragma unroll
        for (int i = 0; i < 4; ++i) {
            int ncol = (nt0 + i) * 16 + (lane & 15);
            float bias = b_comb[ncol];
#pragma unroll
            for (int r = 0; r < 8; ++r) {
                int gm = wg * 64 + mlo + r;
                int b  = gm >> 10, t = gm & 1023;
                __builtin_nontemporal_store(acc[i][r] + bias,
                    ig + ((size_t)t * B_ + b) * G3H + ncol);
            }
        }
    }
}

// ---------------------------------------------------------------------------
// 6) GRU scan. 8 WGs x 512 thr; WG owns 16 batch rows for all 1024 steps.
//    Per step: hg = h_bf16 @ whh^T via WMMA with a one-kt-ahead double-buffered
//    B stream (hides L2 latency at 4 waves/SIMD), ig prefetched into registers
//    during the GEMM phase; then gate math in registers.
// ---------------------------------------------------------------------------
__global__ void __launch_bounds__(512)
k_gru_scan(const float* __restrict__ ig, const uint4* __restrict__ whhp4,
           const float* __restrict__ b_n, float* __restrict__ h_last) {
    extern __shared__ char smem[];
    unsigned short* hbf = (unsigned short*)smem;                 // 16 x 522 bf16
    const unsigned* hbu = (const unsigned*)smem;
    float* hg = (float*)(smem + 16 * XSTR * 4);                  // 16 x 1540 f32

    const int tid  = threadIdx.x;
    const int lane = tid & 31;
    const int wave = tid >> 5;                                   // 0..15
    const int bstart = blockIdx.x * 16;

    float h[16];
#pragma unroll
    for (int m = 0; m < 16; ++m) h[m] = 0.f;
    const float bn = b_n[tid];

    // zero h_bf16 (h0 = 0)
    for (int i = tid; i < 16 * XSTR; i += 512) ((unsigned*)smem)[i] = 0u;
    __syncthreads();

    const int mrow = lane & 15;
    const int khi  = ((lane >> 4) << 3);
    const int m0   = ((lane >> 4) << 3);
    const uint4* bbase = whhp4 + (size_t)(wave * 6) * KT_ * 64 + lane * 2;

    for (int t = 0; t < T_; ++t) {
        // ---- prefetch this step's ig slice (overlaps with the WMMA phase) ----
        const float* igp = ig + ((size_t)t * B_ + bstart) * G3H + tid;
        float igr_[16], igz_[16], ign_[16];
#pragma unroll
        for (int m = 0; m < 16; ++m) {
            igr_[m] = __builtin_nontemporal_load(igp + (size_t)m * G3H);
            igz_[m] = __builtin_nontemporal_load(igp + (size_t)m * G3H + 512);
            ign_[m] = __builtin_nontemporal_load(igp + (size_t)m * G3H + 1024);
        }

        // ---- phase A: hg = h @ w_hh^T, software-pipelined B (1 kt ahead) ----
        v8f acc[6];
#pragma unroll
        for (int i = 0; i < 6; ++i) acc[i] = (v8f){0.f,0.f,0.f,0.f,0.f,0.f,0.f,0.f};

        u32x8 bu[2][6];
#pragma unroll
        for (int i = 0; i < 6; ++i) bu[0][i] = loadB16(bbase + (i * KT_ + 0) * 64);

#pragma unroll
        for (int kt = 0; kt < KT_; ++kt) {
            const int cur = kt & 1, nxt = cur ^ 1;
            if (kt + 1 < KT_) {
#pragma unroll
                for (int i = 0; i < 6; ++i)
                    bu[nxt][i] = loadB16(bbase + (i * KT_ + (kt + 1)) * 64);
            }
            u32x8 au;
#pragma unroll
            for (int v = 0; v < 8; ++v) {
                int k = kt * 32 + ((v & 4) << 2) + khi + ((v & 3) << 1);
                au[v] = hbu[mrow * XSTR + (k >> 1)];
            }
            v16bf A = as16bf(au);
#pragma unroll
            for (int i = 0; i < 6; ++i)
                acc[i] = __builtin_amdgcn_wmma_f32_16x16x32_bf16(
                    false, A, false, as16bf(bu[cur][i]), (short)0, acc[i], false, false);
        }
#pragma unroll
        for (int i = 0; i < 6; ++i) {
            int ncol = (wave * 6 + i) * 16 + (lane & 15);
#pragma unroll
            for (int r = 0; r < 8; ++r) hg[(m0 + r) * HGSTR + ncol] = acc[i][r];
        }
        __syncthreads();

        // ---- phase B: gate math; thread owns column tid of all 16 rows ----
#pragma unroll
        for (int m = 0; m < 16; ++m) {
            float r  = fsigmoid(igr_[m] + hg[m * HGSTR + tid]);
            float z  = fsigmoid(igz_[m] + hg[m * HGSTR + 512 + tid]);
            float nn = ftanh   (ign_[m] + r * (hg[m * HGSTR + 1024 + tid] + bn));
            h[m] = nn + z * (h[m] - nn);
            hbf[m * 522 + tid] = f2bf(h[m]);
        }
        __syncthreads();
    }

#pragma unroll
    for (int m = 0; m < 16; ++m) h_last[(size_t)(bstart + m) * HID + tid] = h[m];
}

// ---------------------------------------------------------------------------
// 7) decoder + split + softplus.  block = batch row, 256 threads = out cols.
// ---------------------------------------------------------------------------
__global__ void k_decoder(const float* __restrict__ h_last, const float* __restrict__ dec_w,
                          const float* __restrict__ dec_b, float* __restrict__ out) {
    __shared__ float hs[HID];
    const int b = blockIdx.x, c = threadIdx.x;
    for (int k = c; k < HID; k += 256) hs[k] = h_last[(size_t)b * HID + k];
    __syncthreads();
    const float* wr = dec_w + (size_t)c * HID;
    float s = dec_b[c];
    for (int k = 0; k < HID; ++k) s += hs[k] * wr[k];
    if (c < CTX) {
        out[(size_t)b * CTX + c] = s;                                   // mu
    } else {
        float sp = (s > 20.0f) ? s : log1pf(expf(s));
        out[(size_t)B_ * CTX + (size_t)b * CTX + (c - CTX)] = sp + 0.1f; // stddev
    }
}

// ---------------------------------------------------------------------------
// Launch
// ---------------------------------------------------------------------------
extern "C" void kernel_launch(void* const* d_in, const int* in_sizes, int n_in,
                              void* d_out, int out_size, void* d_ws, size_t ws_size,
                              hipStream_t stream) {
    const float* obs   = (const float*)d_in[0];
    const float* rew   = (const float*)d_in[1];
    const float* cst   = (const float*)d_in[2];
    const float* ter   = (const float*)d_in[3];
    const float* act   = (const float*)d_in[4];
    const float* enc_w = (const float*)d_in[5];
    const float* enc_b = (const float*)d_in[6];
    const float* w_ih  = (const float*)d_in[7];
    const float* w_hh  = (const float*)d_in[8];
    const float* b_ih  = (const float*)d_in[9];
    const float* b_n   = (const float*)d_in[10];
    const float* dec_w = (const float*)d_in[11];
    const float* dec_b = (const float*)d_in[12];
    (void)in_sizes; (void)n_in; (void)out_size; (void)ws_size;

    // workspace carve (bytes)
    char* ws = (char*)d_ws;
    float*    enc_wT = (float*)   (ws + 0);                    //  1,048,576
    unsigned* wcombp = (unsigned*)(ws + 1048576);              //  1,572,864
    unsigned* whhp   = (unsigned*)(ws + 2621440);              //  1,572,864
    float*    b_comb = (float*)   (ws + 4194304);              //      6,144
    float*    h_last = (float*)   (ws + 4200448);              //    262,144
    float*    ig     = (float*)   (ws + 4462592);              // 805,306,368

    k_transpose_encw<<<512, 512, 0, stream>>>(enc_w, enc_wT);
    k_pack_wcomb   <<<1536, 256, 0, stream>>>(w_ih, enc_wT, wcombp);
    k_pack_whh     <<<1536, 256, 0, stream>>>(w_hh, whhp);
    k_bias_comb    <<<6, 256, 0, stream>>>(w_ih, enc_b, b_ih, b_comb);

    k_ig_gemm<<<2048, 512, 64 * XSTR * 4, stream>>>(
        obs, rew, cst, ter, act, (const uint4*)wcombp, b_comb, ig);

    k_gru_scan<<<8, 512, 16 * XSTR * 4 + 16 * HGSTR * 4, stream>>>(
        ig, (const uint4*)whhp, b_n, h_last);

    k_decoder<<<B_, 256, 0, stream>>>(h_last, dec_w, dec_b, (float*)d_out);
}
